// NT_Xent_37452114821332
// MI455X (gfx1250) — compile-verified, matrix operand order checked by reading, not run on hardware
//
#include <hip/hip_runtime.h>

// Problem constants (match the reference)
#define NTOT  8192      // 2 * BATCH_SIZE
#define BATCH 4096
#define DIM   256
static __device__ constexpr float INV_T  = 10.0f;                 // 1 / 0.1
static __device__ constexpr float K2     = 14.4269504088896341f;  // INV_T * log2(e)
static __device__ constexpr float SMAX2  = 14.4269504088896341f;  // 10 * log2(e) (max logit in base-2)
static __device__ constexpr float LN2    = 0.69314718055994531f;

typedef __attribute__((ext_vector_type(16))) _Float16 v16h;
typedef __attribute__((ext_vector_type(8)))  _Float16 half8;
typedef __attribute__((ext_vector_type(8)))  float    v8f;

// ---------------------------------------------------------------------------
// Kernel 1: row L2-normalize, f32 -> f16.  One wave (32 lanes) per row.
// ---------------------------------------------------------------------------
__global__ __launch_bounds__(256) void normalize_rows(const float* __restrict__ z,
                                                      _Float16* __restrict__ zn) {
  const int wave = threadIdx.x >> 5;
  const int lane = threadIdx.x & 31;
  const int row  = blockIdx.x * 8 + wave;

  const float* p = z + (size_t)row * DIM + lane * 8;
  float4 a = *(const float4*)p;
  float4 b = *(const float4*)(p + 4);
  float ss = a.x*a.x + a.y*a.y + a.z*a.z + a.w*a.w
           + b.x*b.x + b.y*b.y + b.z*b.z + b.w*b.w;
#pragma unroll
  for (int m = 1; m < 32; m <<= 1) ss += __shfl_xor(ss, m, 32);
  const float inv = rsqrtf(ss);

  half8 o;
  o[0] = (_Float16)(a.x * inv); o[1] = (_Float16)(a.y * inv);
  o[2] = (_Float16)(a.z * inv); o[3] = (_Float16)(a.w * inv);
  o[4] = (_Float16)(b.x * inv); o[5] = (_Float16)(b.y * inv);
  o[6] = (_Float16)(b.z * inv); o[7] = (_Float16)(b.w * inv);
  *(half8*)(zn + (size_t)row * DIM + lane * 8) = o;
}

// ---------------------------------------------------------------------------
// WMMA fragment loader for a 16x32 f16 tile of a row-major [.,256] matrix.
// ISA layout (16-bit A 16x32): lane<16 -> row=lane,   K in {k0..k0+7, k0+16..k0+23}
//                              lane>=16 -> row=lane-16, K in {k0+8..k0+15, k0+24..k0+31}
// For C = A * B^T with both operands row-major, the B fragment uses the same
// per-lane pattern (lane -> column index).
// ---------------------------------------------------------------------------
__device__ __forceinline__ v16h load_frag(const _Float16* __restrict__ zn,
                                          int row0, int k0, int lane) {
  const int r = lane & 15;
  const int h = (lane >> 4) & 1;
  const _Float16* p = zn + (size_t)(row0 + r) * DIM + k0 + h * 8;
  union { v16h v; half8 h8[2]; } u;
  u.h8[0] = *(const half8*)p;          // K = k0 + h*8      .. +7   (16B load)
  u.h8[1] = *(const half8*)(p + 16);   // K = k0 + 16 + h*8 .. +7   (16B load)
  return u.v;
}

// ---------------------------------------------------------------------------
// Kernel 2: fused zn*zn^T GEMM + masked fixed-max logsumexp + positive capture.
// Block = 8 waves, each wave owns 16 rows (A panel resident in 64 VGPRs).
// Per column tile: one clause of 16 b128 loads (forced ahead of the WMMAs by
// a sched_barrier so the waitcnt pass emits incremental waits), then 8
// back-to-back WMMAs.  Diagonal / positive-pair masking is hoisted into
// uniform branches taken in exactly 1 of 512 tiles each.
// ---------------------------------------------------------------------------
__global__ __launch_bounds__(256, 1) void ntxent_main(const _Float16* __restrict__ zn,
                                                      float* __restrict__ partials) {
  const int wave    = threadIdx.x >> 5;
  const int lane    = threadIdx.x & 31;
  const int rowBase = blockIdx.x * 128 + wave * 16;
  const int h       = lane >> 4;       // which half-group of lanes
  const int cLane   = lane & 15;       // column within the 16-wide tile

  // Resident A fragments: 16 rows x full K=256 (8 chunks of 32).
  v16h afrag[8];
#pragma unroll
  for (int kk = 0; kk < 8; ++kk) afrag[kk] = load_frag(zn, rowBase, kk * 32, lane);

  // Per-(lane, slot) partial state.  Slot v <-> row (rowBase + v + 8*h).
  float sume[8], posv[8];
#pragma unroll
  for (int v = 0; v < 8; ++v) { sume[v] = 0.0f; posv[v] = 0.0f; }

  for (int c = 0; c < NTOT; c += 16) {
    // Stage the full B tile (16 cols x K=256) in registers.
    v16h bf[8];
#pragma unroll
    for (int kk = 0; kk < 8; ++kk) bf[kk] = load_frag(zn, c, kk * 32, lane);

    // Keep all 16 loads issued before the first WMMA: loads return in order,
    // so WMMA kk only waits for loads 2kk..2kk+1 while the rest stay in flight.
    __builtin_amdgcn_sched_barrier(0);

    v8f acc = {0.f, 0.f, 0.f, 0.f, 0.f, 0.f, 0.f, 0.f};
#pragma unroll
    for (int kk = 0; kk < 8; ++kk) {
      acc = __builtin_amdgcn_wmma_f32_16x16x32_f16(
          /*neg_a=*/false, afrag[kk], /*neg_b=*/false, bf[kk],
          /*c_mod=*/(short)0, acc, /*reuse_a=*/false, /*reuse_b=*/false);
    }

    // Hot path: exp(s - 10) = exp2(acc*K2 - SMAX2); args in [-28.9, 0] so the
    // raw v_exp_f32 needs no denormal fixup.
    float e[8];
#pragma unroll
    for (int v = 0; v < 8; ++v)
      e[v] = __builtin_amdgcn_exp2f(fmaf(acc[v], K2, -SMAX2));

    // Diagonal lives only in tile c == rowBase (uniform scalar branch).
    if (__builtin_expect(c == rowBase, 0)) {
#pragma unroll
      for (int v = 0; v < 8; ++v) e[v] = (cLane == v + 8 * h) ? 0.0f : e[v];
    }
    // Positive pair (col = i ^ BATCH) lives only in tile c == rowBase ^ BATCH.
    if (__builtin_expect(c == (rowBase ^ BATCH), 0)) {
#pragma unroll
      for (int v = 0; v < 8; ++v)
        posv[v] = (cLane == v + 8 * h) ? acc[v] : posv[v];
    }

#pragma unroll
    for (int v = 0; v < 8; ++v) sume[v] += e[v];
  }

  // Merge the 16 lanes that share each row (xor-shuffle tree, width 16).
#pragma unroll
  for (int m = 1; m < 16; m <<= 1) {
#pragma unroll
    for (int v = 0; v < 8; ++v) {
      sume[v] += __shfl_xor(sume[v], m, 16);
      posv[v] += __shfl_xor(posv[v], m, 16);
    }
  }

  __shared__ float red[16];
  if (cLane == 0) {           // lanes 0 and 16 of each wave: 8 rows each
    float loc = 0.0f;
#pragma unroll
    for (int v = 0; v < 8; ++v) {
      // lse = smax + ln(sum) = 10 + ln2 * log2(sum)
      const float lse = 10.0f + LN2 * __builtin_amdgcn_logf(sume[v]);
      loc += lse - posv[v] * INV_T;
    }
    red[wave * 2 + h] = loc;
  }
  __syncthreads();
  if (threadIdx.x == 0) {
    float t = 0.0f;
#pragma unroll
    for (int k = 0; k < 16; ++k) t += red[k];
    partials[blockIdx.x] = t;
  }
}

// ---------------------------------------------------------------------------
// Kernel 3: deterministic ordered final reduction.
// ---------------------------------------------------------------------------
__global__ void ntxent_finalize(const float* __restrict__ partials,
                                float* __restrict__ out) {
  if (threadIdx.x == 0 && blockIdx.x == 0) {
    float t = 0.0f;
    for (int k = 0; k < 64; ++k) t += partials[k];
    out[0] = t / (float)NTOT;
  }
}

// ---------------------------------------------------------------------------
extern "C" void kernel_launch(void* const* d_in, const int* in_sizes, int n_in,
                              void* d_out, int out_size, void* d_ws, size_t ws_size,
                              hipStream_t stream) {
  (void)in_sizes; (void)n_in; (void)out_size; (void)ws_size;
  const float* z = (const float*)d_in[0];

  _Float16* zn16     = (_Float16*)d_ws;                                  // 4 MB
  float*    partials = (float*)((char*)d_ws + (size_t)NTOT * DIM * 2);   // 64 floats

  normalize_rows<<<NTOT / 8, 256, 0, stream>>>(z, zn16);
  ntxent_main<<<NTOT / 128, 256, 0, stream>>>(zn16, partials);
  ntxent_finalize<<<1, 32, 0, stream>>>(partials, (float*)d_out);
}
